// ReweightVLAD_22411139350721
// MI455X (gfx1250) — matrix operand order repaired and checked by reading, not compiled, not run on hardware
//
#include <hip/hip_runtime.h>

// ReweightVLAD for MI455X (gfx1250): CRN folded into a GEMM, all matmuls on
// v_wmma_f32_16x16x32_bf16 (bf16 operands, f32 accumulate). Operands are
// stored pre-swizzled in WMMA fragment order so GEMM loops are b128 loads +
// wmma only. Deterministic (fixed-order K-split reduction, no float atomics).

typedef __attribute__((ext_vector_type(16))) __bf16        v16bf;
typedef __attribute__((ext_vector_type(8)))  float         v8f;
typedef __attribute__((ext_vector_type(8)))  unsigned int  v8u;

#define DEV static __device__ __forceinline__

DEV unsigned short f2bf(float f) {
  unsigned int u = __float_as_uint(f);
  u += 0x7FFFu + ((u >> 16) & 1u);     // round-to-nearest-even
  return (unsigned short)(u >> 16);
}
DEV unsigned int pack2bf(float a, float b) {
  return (unsigned int)f2bf(a) | ((unsigned int)f2bf(b) << 16);
}

// align_corners=True bilinear weight M20[i][q], 7 -> 20 (i in [0,20), q in [0,7))
DEV float m20w(int i, int q) {
  if (i < 0 || i >= 20) return 0.0f;   // zero padding outside the 20x20 grid
  float pos = (float)i * (6.0f / 19.0f);
  int lo = (int)pos; if (lo > 6) lo = 6;
  float frac = pos - (float)lo;
  int hi = lo + 1; if (hi > 6) hi = 6;
  float w = 0.0f;
  if (q == lo) w += 1.0f - frac;
  if (q == hi) w += frac;
  return w;
}

// Load one 16x32 bf16 WMMA operand fragment: lane's 8 dwords are contiguous.
DEV v16bf ldfrag(const unsigned int* base, int lane) {
  const v8u* p = (const v8u*)(base + lane * 8);
  v8u r = *p;
  return __builtin_bit_cast(v16bf, r);
}

// ---------------------------------------------------------------- k_wacc
// Fold acc_w (1x1 conv) into each conv's weights; compute scalar bias term.
__global__ void k_wacc(const float* w_s, const float* w_m, const float* w_l,
                       const float* w_t, const float* b_s, const float* b_m,
                       const float* b_l, const float* b_t, const float* acc_w,
                       const float* acc_b, float* wacc_s, float* wacc_m,
                       float* wacc_l, float* wacc_t, float* cconst) {
  int tid = blockIdx.x * blockDim.x + threadIdx.x;
  const int NS = 512 * 9, NM = 512 * 49, NL = 512 * 81, NT = 512 * 64;
  if (tid < NS) {
    float s = 0.f; for (int o = 0; o < 32; ++o) s += acc_w[o] * w_s[o * NS + tid];
    wacc_s[tid] = s;
  } else if (tid < NS + NM) {
    int i = tid - NS;
    float s = 0.f; for (int o = 0; o < 32; ++o) s += acc_w[32 + o] * w_m[o * NM + i];
    wacc_m[i] = s;
  } else if (tid < NS + NM + NL) {
    int i = tid - NS - NM;
    float s = 0.f; for (int o = 0; o < 20; ++o) s += acc_w[64 + o] * w_l[o * NL + i];
    wacc_l[i] = s;
  } else if (tid < NS + NM + NL + NT) {
    int i = tid - NS - NM - NL;
    float s = 0.f; for (int o = 0; o < 16; ++o) s += acc_w[84 + o] * w_t[o * NT + i];
    wacc_t[i] = s;
  } else if (tid == NS + NM + NL + NT) {
    float s = acc_b[0];
    for (int o = 0; o < 32; ++o) s += acc_w[o] * b_s[o] + acc_w[32 + o] * b_m[o];
    for (int o = 0; o < 20; ++o) s += acc_w[64 + o] * b_l[o];
    for (int o = 0; o < 16; ++o) s += acc_w[84 + o] * b_t[o];
    cconst[0] = s;
  }
}

// ---------------------------------------------------------------- k_build_AT
// Build the folded CRN matrices A,T [cq=25088 x p=64pad] as bf16 in WMMA
// B-operand fragment order (pairs (cq, cq+1) packed per dword).
__global__ void k_build_AT(const float* wacc_s, const float* wacc_m,
                           const float* wacc_l, const float* wacc_t,
                           unsigned int* Amat, unsigned int* Tmat) {
  int tid = blockIdx.x * blockDim.x + threadIdx.x;
  const int ND = 784 * 4 * 256;        // 802816 dwords per matrix
  if (tid >= 2 * ND) return;
  int sel = tid >= ND;                 // 0 = A (x path), 1 = T (t path)
  int d = sel ? tid - ND : tid;
  int ki = d >> 10, ni = (d >> 8) & 3, idx = d & 255;
  int lane = idx >> 3, v = idx & 7;
  int kbase = ki * 32 + ((lane & 16) ? 16 : 0) + 2 * v;   // B-layout K index
  int p = ni * 16 + (lane & 15);
  float vals[2] = {0.f, 0.f};
  if (p < 49) {
    int py = p / 7, px = p % 7;
    for (int h = 0; h < 2; ++h) {
      int cq = kbase + h;
      int c = cq / 49, q = cq % 49;
      int qy = q / 7, qx = q % 7;
      float acc = 0.f;
      if (!sel) {
        { // xs: k=3 stride=3 pad=1
          float ay[3], bx[3];
          for (int j = 0; j < 3; ++j) { ay[j] = m20w(3*py - 1 + j, qy); bx[j] = m20w(3*px - 1 + j, qx); }
          const float* wc = wacc_s + c * 9;
          for (int ky = 0; ky < 3; ++ky) for (int kx = 0; kx < 3; ++kx)
            acc += wc[ky*3 + kx] * ay[ky] * bx[kx];
        }
        { // xm: k=7 stride=5 pad=9
          float ay[7], bx[7];
          for (int j = 0; j < 7; ++j) { ay[j] = m20w(5*py - 9 + j, qy); bx[j] = m20w(5*px - 9 + j, qx); }
          const float* wc = wacc_m + c * 49;
          for (int ky = 0; ky < 7; ++ky) for (int kx = 0; kx < 7; ++kx)
            acc += wc[ky*7 + kx] * ay[ky] * bx[kx];
        }
        { // xl: k=9 stride=2 pad=1
          float ay[9], bx[9];
          for (int j = 0; j < 9; ++j) { ay[j] = m20w(2*py - 1 + j, qy); bx[j] = m20w(2*px - 1 + j, qx); }
          const float* wc = wacc_l + c * 81;
          for (int ky = 0; ky < 9; ++ky) for (int kx = 0; kx < 9; ++kx)
            acc += wc[ky*9 + kx] * ay[ky] * bx[kx];
        }
      } else { // t path: k=8 stride=2 pad=0
        float ay[8], bx[8];
        for (int j = 0; j < 8; ++j) { ay[j] = m20w(2*py + j, qy); bx[j] = m20w(2*px + j, qx); }
        const float* wc = wacc_t + c * 64;
        for (int ky = 0; ky < 8; ++ky) for (int kx = 0; kx < 8; ++kx)
          acc += wc[ky*8 + kx] * ay[ky] * bx[kx];
      }
      vals[h] = acc;
    }
  }
  (sel ? Tmat : Amat)[d] = pack2bf(vals[0], vals[1]);
}

// ---------------------------------------------------------------- k_ln_stats
__global__ void k_ln_stats(const float* x, float* mu, float* rstd) {
  int tid = blockIdx.x * blockDim.x + threadIdx.x;
  if (tid >= 512 * 512) return;
  const float* p = x + (size_t)tid * 49;
  float s = 0.f, ss = 0.f;
  for (int q = 0; q < 49; ++q) { float v = p[q]; s += v; ss += v * v; }
  float m = s * (1.0f / 49.0f);
  float var = ss * (1.0f / 49.0f) - m * m;
  mu[tid] = m;
  rstd[tid] = rsqrtf(var + 1e-5f);
}

// ---------------------------------------------------------------- k_xnswz
// Swizzle LN(x) and raw t into bf16 WMMA A-operand fragments [n-tile][cq-tile].
__global__ void k_xnswz(const float* x, const float* t, const float* mu,
                        const float* rstd, const float* ln_w, const float* ln_b,
                        unsigned int* xnA, unsigned int* tA) {
  int tid = blockIdx.x * blockDim.x + threadIdx.x;
  const int ND = 32 * 784 * 256;       // 6,422,528 dwords per matrix
  if (tid >= 2 * ND) return;
  int sel = tid >= ND;
  int d = sel ? tid - ND : tid;
  int mi = d / (784 * 256);
  int rem = d % (784 * 256);
  int ki = rem >> 8, idx = rem & 255;
  int lane = idx >> 3, v = idx & 7;
  int m = mi * 16 + (lane & 15);
  int klo = (lane & 16) ? 16 : 0;                         // A-layout (16x32 bf16), contiguous k = 16h + s
  int kk = klo + 2 * v;
  int cq = ki * 32 + kk;
  float vals[2];
  for (int h = 0; h < 2; ++h) {
    int c = (cq + h) / 49, q = (cq + h) % 49;
    size_t xi = ((size_t)m * 512 + c) * 49 + q;
    if (!sel) {
      int si = m * 512 + c;
      vals[h] = (x[xi] - mu[si]) * rstd[si] * ln_w[q] + ln_b[q];
    } else {
      vals[h] = t[xi];
    }
  }
  (sel ? tA : xnA)[d] = pack2bf(vals[0], vals[1]);
}

// ---------------------------------------------------------------- k_cw_gemm
// cw[n,p] partials: D[16x16] += xn/t A-frags x A/T B-frags, K split 16 ways.
__global__ void __launch_bounds__(32)
k_cw_gemm(const unsigned int* xnA, const unsigned int* tA,
          const unsigned int* Amat, const unsigned int* Tmat, float* cwpart) {
  int bid = blockIdx.x;                // 2048 = 32(mi) * 4(pi) * 16(chunk)
  int lane = threadIdx.x;
  int chunk = bid & 15, pi = (bid >> 4) & 3, mi = bid >> 6;
  v8f acc = {};
  int kt0 = chunk * 98;                // 1568 combined k-tiles (784 x-path + 784 t-path)
  for (int kt = kt0; kt < kt0 + 98; ++kt) {
    const unsigned int *ap, *bp;
    if (kt < 784) {
      ap = xnA + ((size_t)mi * 784 + kt) * 256;
      bp = Amat + ((size_t)kt * 4 + pi) * 256;
    } else {
      int k2 = kt - 784;
      ap = tA + ((size_t)mi * 784 + k2) * 256;
      bp = Tmat + ((size_t)k2 * 4 + pi) * 256;
    }
    __builtin_prefetch(ap + 256, 0, 1);       // next k-tile (global_prefetch_b8)
    __builtin_prefetch(bp + 4 * 256, 0, 1);
    v16bf a = ldfrag(ap, lane);
    v16bf b = ldfrag(bp, lane);
    acc = __builtin_amdgcn_wmma_f32_16x16x32_bf16(false, a, false, b,
                                                  (short)0, acc, false, false);
  }
  float* outp = cwpart + (((size_t)chunk * 128) + (mi * 4 + pi)) * 256 + lane * 8;
#pragma unroll
  for (int r = 0; r < 8; ++r) outp[r] = acc[r];
}

// ---------------------------------------------------------------- k_cw_reduce
__global__ void k_cw_reduce(const float* cwpart, const float* cconst, float* cw) {
  int tid = blockIdx.x * blockDim.x + threadIdx.x;
  if (tid >= 128 * 256) return;
  int tile = tid >> 8, idx = tid & 255;
  float s = cconst[0];
  for (int ch = 0; ch < 16; ++ch) s += cwpart[((size_t)ch * 128 + tile) * 256 + idx];
  int lane = idx >> 3, r = idx & 7;
  int mi = tile >> 2, pi = tile & 3;
  int n = mi * 16 + r + ((lane & 16) ? 8 : 0);   // f32 C/D 16x16 layout
  int p = pi * 16 + (lane & 15);
  cw[n * 64 + p] = s;
}

// ---------------------------------------------------------------- k_convw_swz
// conv_w [64,512] -> bf16 A-operand fragments (4 m-tiles x 16 k-tiles).
__global__ void k_convw_swz(const float* conv_w, unsigned int* convwA) {
  int tid = blockIdx.x * blockDim.x + threadIdx.x;
  if (tid >= 16384) return;
  int tile = tid >> 8, idx = tid & 255;
  int mi = tile >> 4, ki = tile & 15;
  int lane = idx >> 3, v = idx & 7;
  int m = mi * 16 + (lane & 15);
  int klo = (lane & 16) ? 16 : 0;
  int kk = klo + 2 * v;
  int k = ki * 32 + kk;
  convwA[tid] = pack2bf(conv_w[m * 512 + k], conv_w[m * 512 + k + 1]);
}

// ---------------------------------------------------------------- k_invnrm
__global__ void k_invnrm(const float* x, float* invnrm) {
  int tid = blockIdx.x * blockDim.x + threadIdx.x;
  if (tid >= 512 * 49) return;
  int n = tid / 49, s = tid % 49;
  const float* p = x + (size_t)n * 512 * 49 + s;
  float ss = 0.f;
  for (int c = 0; c < 512; ++c) { float v = p[c * 49]; ss += v * v; }
  invnrm[tid] = 1.0f / fmaxf(sqrtf(ss), 1e-12f);
}

// ---------------------------------------------------------------- k_xf_swz
// L2-normalized x in two bf16 B-operand layouts: [c][s] (logits) and [s][c] (vlad).
__global__ void k_xf_swz(const float* x, const float* invnrm,
                         unsigned int* logitsB, unsigned int* vladB) {
  int tid = blockIdx.x * blockDim.x + threadIdx.x;
  const int ND = 512 * 16384;          // 8,388,608 dwords each
  if (tid >= 2 * ND) return;
  int sel = tid >= ND;
  int d = sel ? tid - ND : tid;
  int n = d >> 14;
  int rem = d & 16383;
  int tile = rem >> 8, idx = rem & 255;
  int lane = idx >> 3, v = idx & 7;
  unsigned int dw;
  if (!sel) {                          // B = xf[c][s], K=c (16 tiles), N=s (4 tiles)
    int ki = tile >> 2, ni = tile & 3;
    int c = ki * 32 + ((lane & 16) ? 16 : 0) + 2 * v;
    int s = ni * 16 + (lane & 15);
    float v0 = 0.f, v1 = 0.f;
    if (s < 49) {
      float inv = invnrm[n * 49 + s];
      v0 = x[((size_t)n * 512 + c) * 49 + s] * inv;
      v1 = x[((size_t)n * 512 + c + 1) * 49 + s] * inv;
    }
    dw = pack2bf(v0, v1);
  } else {                             // B = xf^T[s][c], K=s (2 tiles), N=c (32 tiles)
    int ki = tile >> 5, ni = tile & 31;
    int s0 = ki * 32 + ((lane & 16) ? 16 : 0) + 2 * v;
    int c = ni * 16 + (lane & 15);
    float v0 = 0.f, v1 = 0.f;
    if (s0 < 49)     v0 = x[((size_t)n * 512 + c) * 49 + s0] * invnrm[n * 49 + s0];
    if (s0 + 1 < 49) v1 = x[((size_t)n * 512 + c) * 49 + s0 + 1] * invnrm[n * 49 + s0 + 1];
    dw = pack2bf(v0, v1);
  }
  (sel ? vladB : logitsB)[d] = dw;
}

// ---------------------------------------------------------------- k_logits
__global__ void __launch_bounds__(32)
k_logits(const unsigned int* convwA, const unsigned int* logitsB,
         const float* conv_b, float* logits) {
  int bid = blockIdx.x;                // 8192 = 512(n) * 4(mi) * 4(ni)
  int lane = threadIdx.x;
  int ni = bid & 3, mi = (bid >> 2) & 3, n = bid >> 4;
  const unsigned int* bbase = logitsB + (size_t)n * 16384;
  v8f acc = {};
  for (int kt = 0; kt < 16; ++kt) {
    v16bf a = ldfrag(convwA + (mi * 16 + kt) * 256, lane);
    v16bf b = ldfrag(bbase + (kt * 4 + ni) * 256, lane);
    acc = __builtin_amdgcn_wmma_f32_16x16x32_bf16(false, a, false, b,
                                                  (short)0, acc, false, false);
  }
  int s = ni * 16 + (lane & 15);
  int mh = (lane & 16) ? 8 : 0;
#pragma unroll
  for (int r = 0; r < 8; ++r) {
    int k = mi * 16 + mh + r;
    logits[((size_t)n * 64 + k) * 64 + s] = acc[r] + conv_b[k];
  }
}

// ---------------------------------------------------------------- k_softmax
__global__ void k_softmax(float* logits, const float* cw, float* wbuf) {
  int tid = blockIdx.x * blockDim.x + threadIdx.x;
  if (tid >= 512 * 49) return;
  int n = tid / 49, s = tid % 49;
  float* lp = logits + (size_t)n * 64 * 64 + s;
  float mx = -3.0e38f;
  for (int k = 0; k < 64; ++k) mx = fmaxf(mx, lp[k * 64]);
  float sum = 0.f;
  for (int k = 0; k < 64; ++k) { float e = expf(lp[k * 64] - mx); lp[k * 64] = e; sum += e; }
  float scale = cw[n * 64 + s] / sum;
  for (int k = 0; k < 64; ++k) wbuf[((size_t)n * 64 + k) * 49 + s] = lp[k * 64] * scale;
}

// ---------------------------------------------------------------- k_w_swz
// w [n][64][49] -> bf16 A-operand fragments (4 m-tiles x 2 k-tiles per n).
__global__ void k_w_swz(const float* wbuf, unsigned int* wA) {
  int tid = blockIdx.x * blockDim.x + threadIdx.x;
  if (tid >= 512 * 2048) return;
  int n = tid >> 11;
  int rem = tid & 2047;
  int tile = rem >> 8, idx = rem & 255;
  int mi = tile >> 1, ki = tile & 1;
  int lane = idx >> 3, v = idx & 7;
  int m = mi * 16 + (lane & 15);
  int klo = (lane & 16) ? 16 : 0;
  int kk = klo + 2 * v;
  int s = ki * 32 + kk;
  float v0 = (s < 49)     ? wbuf[((size_t)n * 64 + m) * 49 + s]     : 0.f;
  float v1 = (s + 1 < 49) ? wbuf[((size_t)n * 64 + m) * 49 + s + 1] : 0.f;
  wA[tid] = pack2bf(v0, v1);
}

// ---------------------------------------------------------------- k_sumw
__global__ void k_sumw(const float* wbuf, float* sumw) {
  int tid = blockIdx.x * blockDim.x + threadIdx.x;
  if (tid >= 512 * 64) return;
  const float* p = wbuf + (size_t)tid * 49;
  float s = 0.f;
  for (int i = 0; i < 49; ++i) s += p[i];
  sumw[tid] = s;
}

// ---------------------------------------------------------------- k_vlad
__global__ void __launch_bounds__(32)
k_vlad(const unsigned int* wA, const unsigned int* vladB, const float* centroids,
       const float* sumw, float* out) {
  int bid = blockIdx.x;                // 65536 = 512(n) * 4(mi) * 32(ni)
  int lane = threadIdx.x;
  int ni = bid & 31, mi = (bid >> 5) & 3, n = bid >> 7;
  const unsigned int* ab = wA + (size_t)n * 2048;
  const unsigned int* bb = vladB + (size_t)n * 16384;
  v8f acc = {};
  for (int kt = 0; kt < 2; ++kt) {
    v16bf a = ldfrag(ab + (mi * 2 + kt) * 256, lane);
    v16bf b = ldfrag(bb + (kt * 32 + ni) * 256, lane);
    acc = __builtin_amdgcn_wmma_f32_16x16x32_bf16(false, a, false, b,
                                                  (short)0, acc, false, false);
  }
  int c = ni * 16 + (lane & 15);
  int mh = (lane & 16) ? 8 : 0;
#pragma unroll
  for (int r = 0; r < 8; ++r) {
    int k = mi * 16 + mh + r;
    out[(size_t)n * 32768 + k * 512 + c] =
        acc[r] - centroids[k * 512 + c] * sumw[n * 64 + k];
  }
}

// ---------------------------------------------------------------- k_intra
__global__ void k_intra(float* out, float* s2) {
  int row = blockIdx.x;                // 32768 = n*64 + k
  int t = threadIdx.x;                 // 64
  float* p = out + (size_t)row * 512;
  float ss = 0.f;
  for (int i = t; i < 512; i += 64) { float v = p[i]; ss += v * v; }
  __shared__ float red[64];
  __shared__ float sc;
  red[t] = ss;
  __syncthreads();
  if (t == 0) {
    float s = 0.f;
    for (int i = 0; i < 64; ++i) s += red[i];
    float scale = 1.0f / fmaxf(sqrtf(s), 1e-12f);
    sc = scale;
    s2[row] = s * scale * scale;
  }
  __syncthreads();
  float scale = sc;
  for (int i = t; i < 512; i += 64) p[i] *= scale;
}

// ---------------------------------------------------------------- k_global
__global__ void k_global(float* out, const float* s2) {
  int n = blockIdx.x;
  int t = threadIdx.x;                 // 256
  __shared__ float sc;
  if (t == 0) {
    float g = 0.f;
    for (int i = 0; i < 64; ++i) g += s2[n * 64 + i];
    sc = 1.0f / fmaxf(sqrtf(g), 1e-12f);
  }
  __syncthreads();
  float scale = sc;
  float* p = out + (size_t)n * 32768;
  for (int i = t; i < 32768; i += 256) p[i] *= scale;
}

// ================================================================ launch
extern "C" void kernel_launch(void* const* d_in, const int* in_sizes, int n_in,
                              void* d_out, int out_size, void* d_ws, size_t ws_size,
                              hipStream_t stream) {
  const float* x         = (const float*)d_in[0];
  const float* t         = (const float*)d_in[1];
  const float* centroids = (const float*)d_in[2];
  const float* conv_w    = (const float*)d_in[3];
  const float* conv_b    = (const float*)d_in[4];
  const float* ln_w      = (const float*)d_in[5];
  const float* ln_b      = (const float*)d_in[6];
  const float* w_t       = (const float*)d_in[7];
  const float* b_t       = (const float*)d_in[8];
  const float* w_s       = (const float*)d_in[9];
  const float* b_s       = (const float*)d_in[10];
  const float* w_m       = (const float*)d_in[11];
  const float* b_m       = (const float*)d_in[12];
  const float* w_l       = (const float*)d_in[13];
  const float* b_l       = (const float*)d_in[14];
  const float* acc_w     = (const float*)d_in[15];
  const float* acc_b     = (const float*)d_in[16];
  float* out = (float*)d_out;

  char* ws = (char*)d_ws;
  size_t off = 0;
  auto alloc = [&](size_t bytes) -> char* {
    char* p = ws + off;
    off = (off + bytes + 255) & ~(size_t)255;
    return p;
  };
  float* wacc_s  = (float*)alloc(4608 * 4);
  float* wacc_m  = (float*)alloc(25088 * 4);
  float* wacc_l  = (float*)alloc(41472 * 4);
  float* wacc_t  = (float*)alloc(32768 * 4);
  float* cconst  = (float*)alloc(4);
  unsigned int* Amat    = (unsigned int*)alloc((size_t)802816 * 4);
  unsigned int* Tmat    = (unsigned int*)alloc((size_t)802816 * 4);
  float* lnmu    = (float*)alloc((size_t)262144 * 4);
  float* lnrstd  = (float*)alloc((size_t)262144 * 4);
  unsigned int* xnA     = (unsigned int*)alloc((size_t)6422528 * 4);
  unsigned int* tA      = (unsigned int*)alloc((size_t)6422528 * 4);
  float* cwpart  = (float*)alloc((size_t)16 * 128 * 256 * 4);
  float* cw      = (float*)alloc((size_t)512 * 64 * 4);
  unsigned int* convwA  = (unsigned int*)alloc((size_t)16384 * 4);
  float* invnrm  = (float*)alloc((size_t)25088 * 4);
  unsigned int* logitsB = (unsigned int*)alloc((size_t)8388608 * 4);
  unsigned int* vladB   = (unsigned int*)alloc((size_t)8388608 * 4);
  float* logits  = (float*)alloc((size_t)512 * 64 * 64 * 4);
  float* wbuf    = (float*)alloc((size_t)512 * 64 * 49 * 4);
  unsigned int* wA      = (unsigned int*)alloc((size_t)1048576 * 4);
  float* sumw    = (float*)alloc((size_t)32768 * 4);
  float* s2      = (float*)alloc((size_t)32768 * 4);
  (void)in_sizes; (void)n_in; (void)out_size; (void)ws_size;

  // CRN weight folding and folded-matrix construction
  k_wacc<<<407, 256, 0, stream>>>(w_s, w_m, w_l, w_t, b_s, b_m, b_l, b_t,
                                  acc_w, acc_b, wacc_s, wacc_m, wacc_l, wacc_t,
                                  cconst);
  k_build_AT<<<6272, 256, 0, stream>>>(wacc_s, wacc_m, wacc_l, wacc_t, Amat, Tmat);

  // LayerNorm + operand swizzles
  k_ln_stats<<<1024, 256, 0, stream>>>(x, lnmu, lnrstd);
  k_xnswz<<<50176, 256, 0, stream>>>(x, t, lnmu, lnrstd, ln_w, ln_b, xnA, tA);

  // cw = folded CRN GEMM (bf16 WMMA, K split 16 ways, fixed-order reduce)
  k_cw_gemm<<<2048, 32, 0, stream>>>(xnA, tA, Amat, Tmat, cwpart);
  k_cw_reduce<<<128, 256, 0, stream>>>(cwpart, cconst, cw);

  // VLAD path operands
  k_convw_swz<<<64, 256, 0, stream>>>(conv_w, convwA);
  k_invnrm<<<98, 256, 0, stream>>>(x, invnrm);
  k_xf_swz<<<65536, 256, 0, stream>>>(x, invnrm, logitsB, vladB);

  // logits GEMM + softmax * cw
  k_logits<<<8192, 32, 0, stream>>>(convwA, logitsB, conv_b, logits);
  k_softmax<<<98, 256, 0, stream>>>(logits, cw, wbuf);
  k_w_swz<<<4096, 256, 0, stream>>>(wbuf, wA);
  k_sumw<<<128, 256, 0, stream>>>(wbuf, sumw);

  // vlad GEMM + centroid correction, then intra / global L2 normalization
  k_vlad<<<65536, 32, 0, stream>>>(wA, vladB, centroids, sumw, out);
  k_intra<<<32768, 64, 0, stream>>>(out, s2);
  k_global<<<512, 256, 0, stream>>>(out, s2);
}
